// RnnTextRepresentationModel_3186865734217
// MI455X (gfx1250) — matrix operand order repaired and checked by reading, not compiled, not run on hardware
//
#include <hip/hip_runtime.h>

// Problem dimensions (match reference)
#define B_   256
#define L_   256
#define H_   512
#define G4H  2048      // 4*H
#define CH_  64
#define K3H  1536      // 3*H

typedef __attribute__((ext_vector_type(16))) _Float16 v16h;
typedef __attribute__((ext_vector_type(8)))  _Float16 v8h;
typedef __attribute__((ext_vector_type(8)))  float    v8f;

__device__ __forceinline__ float sigmoidf_(float x) { return 1.0f / (1.0f + expf(-x)); }
__device__ __forceinline__ float eluf_(float x)     { return x > 0.0f ? x : expm1f(x); }

// A-matrix fragment, 16x32 f16 (MxK). lane l holds row l%16.
// lanes 0-15: K = 0..15; lanes 16-31: K = 16..31.
__device__ __forceinline__ v16h loadA16(const _Float16* __restrict__ rowptr, int lanehi) {
    return *(const v16h*)(rowptr + (lanehi ? 16 : 0));
}

// B-matrix fragment, 32x16 f16 (KxN). lane l holds column l%16 (= one weight row,
// contiguous in K). lanes 0-15: K=0..15; lanes 16-31: K=16..31.
__device__ __forceinline__ v16h loadB16(const _Float16* __restrict__ rowptr, int lanehi) {
    return *(const v16h*)(rowptr + (lanehi ? 16 : 0));
}

// ---------------------------------------------------------------------------
// Fused LSTM step: gates = x_t @ Wih^T + h_prev @ Whh^T + bias, then cell update.
// Grid: 128 blocks x 128 threads = 512 waves. wave -> (mTile in [0,16), nTile in [0,32)).
// Each wave computes the SAME 16x16 (batch x hidden) tile in all 4 gate blocks,
// so the full cell nonlinearity is done in registers.
// ---------------------------------------------------------------------------
__global__ __launch_bounds__(128) void lstm_step_kernel(
    const _Float16* __restrict__ xbase,   // [B, L, H] layer input (f16)
    const _Float16* __restrict__ h_in,    // [B, H]
    _Float16*       __restrict__ h_out,   // [B, H]
    float*          __restrict__ cbuf,    // [B, H] (read-modify-write, tile-owned)
    const float*    __restrict__ bias,    // [4H] = bih + bhh
    const _Float16* __restrict__ Wih,     // [4H, H] row-major f16
    const _Float16* __restrict__ Whh,     // [4H, H] row-major f16
    const int*      __restrict__ lengths, // [B]
    _Float16*       __restrict__ hs_out,  // [B, L, H] or nullptr
    int t)
{
    const int lane   = threadIdx.x & 31;
    const int lanehi = lane >> 4;
    const int wid    = blockIdx.x * (blockDim.x >> 5) + (threadIdx.x >> 5);
    const int mTile  = wid & 15;                 // batch tile
    const int nTile  = wid >> 4;                 // hidden tile (0..31)
    const int arow   = mTile * 16 + (lane & 15); // A-matrix batch row for this lane
    const int n      = nTile * 16 + (lane & 15); // hidden column for this lane

    const _Float16* xrow = xbase + (size_t)(arow * L_ + t) * H_;
    const _Float16* hrow = h_in + (size_t)arow * H_;

    const _Float16* wih_row[4];
    const _Float16* whh_row[4];
#pragma unroll
    for (int g = 0; g < 4; ++g) {
        wih_row[g] = Wih + (size_t)(g * H_ + n) * H_;
        whh_row[g] = Whh + (size_t)(g * H_ + n) * H_;
    }

    v8f acc[4] = {};

    // x_t @ Wih^T  (K = 0..H)
#pragma unroll 4
    for (int k0 = 0; k0 < H_; k0 += 32) {
        v16h a = loadA16(xrow + k0, lanehi);
#pragma unroll
        for (int g = 0; g < 4; ++g) {
            v16h b = loadB16(wih_row[g] + k0, lanehi);
            acc[g] = __builtin_amdgcn_wmma_f32_16x16x32_f16(
                false, a, false, b, (short)0, acc[g], false, false);
        }
    }
    // h_prev @ Whh^T  (K = 0..H)
#pragma unroll 4
    for (int k0 = 0; k0 < H_; k0 += 32) {
        v16h a = loadA16(hrow + k0, lanehi);
#pragma unroll
        for (int g = 0; g < 4; ++g) {
            v16h b = loadB16(whh_row[g] + k0, lanehi);
            acc[g] = __builtin_amdgcn_wmma_f32_16x16x32_f16(
                false, a, false, b, (short)0, acc[g], false, false);
        }
    }

    const float bI = bias[0 * H_ + n];
    const float bF = bias[1 * H_ + n];
    const float bG = bias[2 * H_ + n];
    const float bO = bias[3 * H_ + n];

    // C/D layout: VGPR r, this lane -> (M = mTile*16 + r + 8*lanehi, N = n)
#pragma unroll
    for (int r = 0; r < 8; ++r) {
        const int brow = mTile * 16 + r + lanehi * 8;
        const size_t idx = (size_t)brow * H_ + n;
        const bool valid = t < lengths[brow];
        float ig = sigmoidf_(acc[0][r] + bI);
        float fg = sigmoidf_(acc[1][r] + bF);
        float gg = tanhf(acc[2][r] + bG);
        float og = sigmoidf_(acc[3][r] + bO);
        float cp = cbuf[idx];
        float cn = fg * cp + ig * gg;
        float hn = og * tanhf(cn);
        if (!valid) { cn = cp; hn = (float)h_in[idx]; }
        cbuf[idx]  = cn;
        h_out[idx] = (_Float16)hn;
        if (hs_out) hs_out[(size_t)(brow * L_ + t) * H_ + n] = (_Float16)hn;
    }
}

// ---------------------------------------------------------------------------
// Classifier stage 1: hidden[B*L, 64] = elu(feat @ W1^T + b1), feat = [x | c0 | c1]
// gathered on the fly. One wave per 16-row M tile, 4 N tiles of 16 each (N=64).
// ---------------------------------------------------------------------------
__global__ __launch_bounds__(128) void classifier1_kernel(
    const _Float16* __restrict__ Xh,     // [B*L, H]
    const _Float16* __restrict__ c0h,    // [B, H]
    const _Float16* __restrict__ c1h,    // [B, H]
    const _Float16* __restrict__ W1h,    // [CH, 3H] row-major
    const float*    __restrict__ b1,     // [CH]
    _Float16*       __restrict__ hidden) // [B*L, CH]
{
    const int lane   = threadIdx.x & 31;
    const int lanehi = lane >> 4;
    const int mTile  = blockIdx.x * (blockDim.x >> 5) + (threadIdx.x >> 5); // 0..4095
    const int m      = mTile * 16 + (lane & 15);
    const int bidx   = (mTile * 16) / L_;   // uniform across wave (L%16==0)

    const _Float16* xrow  = Xh  + (size_t)m * H_;
    const _Float16* c0row = c0h + (size_t)bidx * H_;
    const _Float16* c1row = c1h + (size_t)bidx * H_;

    const _Float16* w1row[4];
#pragma unroll
    for (int nt = 0; nt < 4; ++nt)
        w1row[nt] = W1h + (size_t)(nt * 16 + (lane & 15)) * K3H;

    v8f acc[4] = {};

#pragma unroll 2
    for (int k0 = 0; k0 < K3H; k0 += 32) {
        const _Float16* arow =
            (k0 < H_)     ? (xrow  + k0) :
            (k0 < 2 * H_) ? (c0row + (k0 - H_)) :
                            (c1row + (k0 - 2 * H_));
        v16h a = loadA16(arow, lanehi);
#pragma unroll
        for (int nt = 0; nt < 4; ++nt) {
            v16h b = loadB16(w1row[nt] + k0, lanehi);
            acc[nt] = __builtin_amdgcn_wmma_f32_16x16x32_f16(
                false, a, false, b, (short)0, acc[nt], false, false);
        }
    }

#pragma unroll
    for (int nt = 0; nt < 4; ++nt) {
        const int n = nt * 16 + (lane & 15);
        const float bb = b1[n];
#pragma unroll
        for (int r = 0; r < 8; ++r) {
            const int row = mTile * 16 + r + lanehi * 8;
            hidden[(size_t)row * CH_ + n] = (_Float16)eluf_(acc[nt][r] + bb);
        }
    }
}

// Classifier stage 2: out[B*L, 2] = elu(hidden @ W2^T + b2) * mask. Tiny N=2 -> VALU.
__global__ void classifier2_kernel(
    const _Float16* __restrict__ hidden, const float* __restrict__ W2,
    const float* __restrict__ b2, const int* __restrict__ lengths,
    float* __restrict__ out)
{
    const int row = blockIdx.x * blockDim.x + threadIdx.x;
    if (row >= B_ * L_) return;
    const int b = row >> 8, l = row & (L_ - 1);
    float s0 = b2[0], s1 = b2[1];
    const _Float16* h = hidden + (size_t)row * CH_;
#pragma unroll 8
    for (int k = 0; k < CH_; ++k) {
        float hv = (float)h[k];
        s0 += hv * W2[k];
        s1 += hv * W2[CH_ + k];
    }
    const bool valid = l < lengths[b];
    out[row * 2 + 0] = valid ? eluf_(s0) : 0.0f;
    out[row * 2 + 1] = valid ? eluf_(s1) : 0.0f;
}

// ---- small utilities ----
__global__ void cvt_f32_f16_kernel(const float* __restrict__ in, _Float16* __restrict__ out, int n) {
    for (int i = blockIdx.x * blockDim.x + threadIdx.x; i < n; i += gridDim.x * blockDim.x)
        out[i] = (_Float16)in[i];
}
__global__ void addbias_kernel(const float* __restrict__ a, const float* __restrict__ b,
                               float* __restrict__ out, int n) {
    for (int i = blockIdx.x * blockDim.x + threadIdx.x; i < n; i += gridDim.x * blockDim.x)
        out[i] = a[i] + b[i];
}
__global__ void zero_u32_kernel(unsigned int* __restrict__ p, int n) {
    for (int i = blockIdx.x * blockDim.x + threadIdx.x; i < n; i += gridDim.x * blockDim.x)
        p[i] = 0u;
}

extern "C" void kernel_launch(void* const* d_in, const int* in_sizes, int n_in,
                              void* d_out, int out_size, void* d_ws, size_t ws_size,
                              hipStream_t stream) {
    (void)in_sizes; (void)n_in; (void)out_size; (void)ws_size;
    const float* x       = (const float*)d_in[0];
    const int*   lengths = (const int*)d_in[1];
    const float* Wih0 = (const float*)d_in[2];
    const float* Whh0 = (const float*)d_in[3];
    const float* bih0 = (const float*)d_in[4];
    const float* bhh0 = (const float*)d_in[5];
    const float* Wih1 = (const float*)d_in[6];
    const float* Whh1 = (const float*)d_in[7];
    const float* bih1 = (const float*)d_in[8];
    const float* bhh1 = (const float*)d_in[9];
    const float* W1 = (const float*)d_in[10];
    const float* b1 = (const float*)d_in[11];
    const float* W2 = (const float*)d_in[12];
    const float* b2 = (const float*)d_in[13];
    float* out = (float*)d_out;

    // Carve workspace (each region 256B-aligned).
    char* ws = (char*)d_ws;
    size_t off = 0;
    auto carve = [&](size_t bytes) -> char* {
        char* p = ws + off;
        off += (bytes + 255) & ~(size_t)255;
        return p;
    };
    _Float16* Xh    = (_Float16*)carve((size_t)B_ * L_ * H_ * 2);  // 64 MB
    _Float16* hs0   = (_Float16*)carve((size_t)B_ * L_ * H_ * 2);  // 64 MB
    _Float16* Wih0h = (_Float16*)carve((size_t)G4H * H_ * 2);
    _Float16* Whh0h = (_Float16*)carve((size_t)G4H * H_ * 2);
    _Float16* Wih1h = (_Float16*)carve((size_t)G4H * H_ * 2);
    _Float16* Whh1h = (_Float16*)carve((size_t)G4H * H_ * 2);
    _Float16* W1h   = (_Float16*)carve((size_t)CH_ * K3H * 2);
    float*    bias0 = (float*)carve((size_t)G4H * 4);
    float*    bias1 = (float*)carve((size_t)G4H * 4);
    _Float16* hbuf0 = (_Float16*)carve((size_t)B_ * H_ * 2);
    _Float16* hbuf1 = (_Float16*)carve((size_t)B_ * H_ * 2);
    float*    c0    = (float*)carve((size_t)B_ * H_ * 4);
    float*    c1    = (float*)carve((size_t)B_ * H_ * 4);
    _Float16* c0h   = (_Float16*)carve((size_t)B_ * H_ * 2);
    _Float16* c1h   = (_Float16*)carve((size_t)B_ * H_ * 2);
    _Float16* hidden= (_Float16*)carve((size_t)B_ * L_ * CH_ * 2); // 8 MB

    // Precision prep: f32 -> f16 copies of activations & weights; fused biases.
    cvt_f32_f16_kernel<<<4096, 256, 0, stream>>>(x, Xh, B_ * L_ * H_);
    cvt_f32_f16_kernel<<<512, 256, 0, stream>>>(Wih0, Wih0h, G4H * H_);
    cvt_f32_f16_kernel<<<512, 256, 0, stream>>>(Whh0, Whh0h, G4H * H_);
    cvt_f32_f16_kernel<<<512, 256, 0, stream>>>(Wih1, Wih1h, G4H * H_);
    cvt_f32_f16_kernel<<<512, 256, 0, stream>>>(Whh1, Whh1h, G4H * H_);
    cvt_f32_f16_kernel<<<96, 256, 0, stream>>>(W1, W1h, CH_ * K3H);
    addbias_kernel<<<8, 256, 0, stream>>>(bih0, bhh0, bias0, G4H);
    addbias_kernel<<<8, 256, 0, stream>>>(bih1, bhh1, bias1, G4H);

    // ---- Layer 0 ----
    zero_u32_kernel<<<128, 256, 0, stream>>>((unsigned int*)hbuf0, B_ * H_ / 2);
    zero_u32_kernel<<<256, 256, 0, stream>>>((unsigned int*)c0, B_ * H_);
    for (int t = 0; t < L_; ++t) {
        _Float16* hin  = (t & 1) ? hbuf1 : hbuf0;
        _Float16* hout = (t & 1) ? hbuf0 : hbuf1;
        lstm_step_kernel<<<128, 128, 0, stream>>>(
            Xh, hin, hout, c0, bias0, Wih0h, Whh0h, lengths, hs0, t);
    }

    // ---- Layer 1 (input = hs0) ----
    zero_u32_kernel<<<128, 256, 0, stream>>>((unsigned int*)hbuf0, B_ * H_ / 2);
    zero_u32_kernel<<<256, 256, 0, stream>>>((unsigned int*)c1, B_ * H_);
    for (int t = 0; t < L_; ++t) {
        _Float16* hin  = (t & 1) ? hbuf1 : hbuf0;
        _Float16* hout = (t & 1) ? hbuf0 : hbuf1;
        lstm_step_kernel<<<128, 128, 0, stream>>>(
            hs0, hin, hout, c1, bias1, Wih1h, Whh1h, lengths, nullptr, t);
    }

    // Final cell states to f16 for the classifier GEMM.
    cvt_f32_f16_kernel<<<64, 256, 0, stream>>>(c0, c0h, B_ * H_);
    cvt_f32_f16_kernel<<<64, 256, 0, stream>>>(c1, c1h, B_ * H_);

    // Classifier head.
    classifier1_kernel<<<1024, 128, 0, stream>>>(Xh, c0h, c1h, W1h, b1, hidden);
    classifier2_kernel<<<(B_ * L_) / 256, 256, 0, stream>>>(hidden, W2, b2, lengths, out);
}
